// NodeAttention_66365834658168
// MI455X (gfx1250) — matrix operand order repaired and checked by reading, not compiled, not run on hardware
//
#include <hip/hip_runtime.h>
#include <hip/hip_bf16.h>

// ---------------------------------------------------------------------------
// Problem constants (from the reference)
// ---------------------------------------------------------------------------
#define NPG     116              // nodes per graph
#define DHEAD   116
#define HID     256
#define NGRAPH  512
#define NTOT    (NGRAPH * NPG)   // 59392
#define DEG     8
#define DOUT    232              // heads*dhead
#define NCOLS   928              // q|k|v|s concatenated
#define QKV_STRIDE 696           // q,k,v in workspace; s straight to d_out
#define KL_STRIDE  236           // padded f32 LDS row (232+4 -> 16B-aligned rows)

#define TILE_M  64
#define TILE_N  64
#define A_STRIDE 264             // bf16 LDS row for A (256+8 pad -> bank-tiled b128)
#define B_STRIDE 72              // bf16 LDS row for B (64+8 pad)

typedef __bf16 v4bf  __attribute__((ext_vector_type(4)));
typedef __bf16 v8bf  __attribute__((ext_vector_type(8)));
typedef __bf16 v16bf __attribute__((ext_vector_type(16)));
typedef float  v8f   __attribute__((ext_vector_type(8)));

// ---------------------------------------------------------------------------
// CDNA5 async global->LDS copy (ASYNCcnt-tracked, bypasses VGPRs)
// ---------------------------------------------------------------------------
__device__ __forceinline__ uint32_t lds_addr(const void* p) {
    return (uint32_t)(uintptr_t)(__attribute__((address_space(3))) const void*)p;
}
__device__ __forceinline__ void async_g2l_b128(uint32_t lds_off, const void* g) {
    asm volatile("global_load_async_to_lds_b128 %0, %1, off"
                 :: "v"(lds_off), "v"(g) : "memory");
}
__device__ __forceinline__ void wait_async0() {
    asm volatile("s_wait_asynccnt 0x0" ::: "memory");
}

// ---------------------------------------------------------------------------
// Prep 1: h (f32) -> h_bf (bf16), done once so 15 column-blocks re-read 512B
// rows instead of 1KB + re-converting.  Grid: NTOT/4 blocks x 256.
// ---------------------------------------------------------------------------
__global__ void __launch_bounds__(256)
cvt_h_kernel(const float* __restrict__ h, __bf16* __restrict__ hb)
{
    size_t i = (size_t)blockIdx.x * 256 + threadIdx.x;   // float4 slot
    float4 v = ((const float4*)h)[i];
    v4bf b = { (__bf16)v.x, (__bf16)v.y, (__bf16)v.z, (__bf16)v.w };
    *(v4bf*)(hb + i * 4) = b;
}

// ---------------------------------------------------------------------------
// Prep 2: pack [Wq|Wk|Wv|Ws] -> Wb[256][928] bf16 row-major. Grid: 256 x 256.
// ---------------------------------------------------------------------------
__global__ void __launch_bounds__(256)
pack_w_kernel(const float* __restrict__ Wq, const float* __restrict__ Wk,
              const float* __restrict__ Wv, const float* __restrict__ Ws,
              __bf16* __restrict__ Wb)
{
    int k = blockIdx.x;
    for (int n = threadIdx.x; n < NCOLS; n += 256) {
        const float* Wsel; int c;
        if      (n < 232) { Wsel = Wq; c = n;       }
        else if (n < 464) { Wsel = Wk; c = n - 232; }
        else if (n < 696) { Wsel = Wv; c = n - 464; }
        else              { Wsel = Ws; c = n - 696; }
        Wb[(size_t)k * NCOLS + n] = (__bf16)Wsel[(size_t)k * DOUT + c];
    }
}

// ---------------------------------------------------------------------------
// GEMM: qkvs[:,0:696] = h @ [Wq|Wk|Wv] ; hproj = h @ Ws (skip term, to d_out).
// 256 thr (8 waves). Block tile 64x64, full K=256 in LDS via async copies.
// Wave owns 16(M)x32(N): one A fragment feeds two v_wmma_f32_16x16x32_bf16.
// ---------------------------------------------------------------------------
__global__ void __launch_bounds__(256)
gt_gemm_kernel(const __bf16* __restrict__ hb, const __bf16* __restrict__ Wb,
               float* __restrict__ qkvs, float* __restrict__ hproj)
{
    extern __shared__ __align__(16) char smemg[];
    __bf16* Alds = (__bf16*)smemg;                 // 64 x A_STRIDE
    __bf16* Blds = Alds + TILE_M * A_STRIDE;       // 256 x B_STRIDE

    const int tid   = threadIdx.x;
    const int mbase = blockIdx.y * TILE_M;         // 928 * 64 = 59392
    const int nbase = blockIdx.x * TILE_N;         // 15 blocks (last ragged)
    const uint32_t aoff = lds_addr(Alds);
    const uint32_t boff = lds_addr(Blds);

    // Stage A: 64 rows x 512B -> LDS (528B stride), 16B async chunks
    #pragma unroll
    for (int j = 0; j < 8; ++j) {
        int idx = tid + j * 256;                   // 0..2047
        int r = idx >> 5, c = idx & 31;
        async_g2l_b128(aoff + r * (A_STRIDE * 2) + c * 16,
                       (const char*)(hb + (size_t)(mbase + r) * HID) + c * 16);
    }
    // Stage B: 256 rows x 128B -> LDS (144B stride), guard ragged tail cols
    #pragma unroll
    for (int j = 0; j < 8; ++j) {
        int idx = tid + j * 256;                   // 0..2047
        int k = idx >> 3, c = idx & 7;
        if (nbase + c * 8 < NCOLS)
            async_g2l_b128(boff + k * (B_STRIDE * 2) + c * 16,
                           (const char*)(Wb + (size_t)k * NCOLS + nbase) + c * 16);
    }
    wait_async0();
    __syncthreads();

    const int lane = tid & 31;
    const int wv   = tid >> 5;
    const int mt   = wv & 3;                       // 0..3
    const int nt   = wv >> 2;                      // 0..1
    if (nbase + nt * 32 >= NCOLS) return;          // uniform per wave (last block)

    const int m    = lane & 15;
    const int koff = (lane >> 4) << 3;             // A layout: upper half owns K+8
    const __bf16* Ab    = Alds + (mt * 16 + m) * A_STRIDE + koff;
    const __bf16* Bbase = Blds + nt * 32;

    v8f acc0 = {}, acc1 = {};
    #pragma unroll
    for (int k0 = 0; k0 < HID; k0 += 32) {
        // A 16x32 fragment: elems 0-7 = K[koff..+7], 8-15 = K[koff+16..+23]
        v8bf alo = *(const v8bf*)(Ab + k0);
        v8bf ahi = *(const v8bf*)(Ab + k0 + 16);
        v16bf av = __builtin_shufflevector(alo, ahi,
                        0,1,2,3,4,5,6,7,8,9,10,11,12,13,14,15);
        // B 32x16 fragments: lane&15 = N, elems 0-7 = K[koff..+7], 8-15 = K[koff+16..+23]
        const __bf16* Br = Bbase + (size_t)(k0 + koff) * B_STRIDE + m;
        v16bf b0, b1;
        #pragma unroll
        for (int s = 0; s < 8; ++s) {
            b0[s]     = Br[s * B_STRIDE];
            b0[s + 8] = Br[(s + 16) * B_STRIDE];
            b1[s]     = Br[s * B_STRIDE + 16];
            b1[s + 8] = Br[(s + 16) * B_STRIDE + 16];
        }
        acc0 = __builtin_amdgcn_wmma_f32_16x16x32_bf16(
                   false, av, false, b0, (short)0, acc0, false, false);
        acc1 = __builtin_amdgcn_wmma_f32_16x16x32_bf16(
                   false, av, false, b1, (short)0, acc1, false, false);
    }

    // Epilogue. D layout: VGPR r -> M = r (+8 upper half), N = lane&15.
    const int rowBase = mbase + mt * 16 + ((lane >> 4) << 3);
    #pragma unroll
    for (int t = 0; t < 2; ++t) {
        const v8f acc = t ? acc1 : acc0;
        const int cb  = nbase + nt * 32 + t * 16;  // uniform per wave
        const int col = cb + m;
        if (cb + 16 <= QKV_STRIDE) {               // fully q|k|v
            #pragma unroll
            for (int r = 0; r < 8; ++r)
                qkvs[(size_t)(rowBase + r) * QKV_STRIDE + col] = acc[r];
        } else if (cb >= QKV_STRIDE) {             // fully skip-term
            #pragma unroll
            for (int r = 0; r < 8; ++r)
                hproj[(size_t)(rowBase + r) * DOUT + (col - QKV_STRIDE)] = acc[r];
        } else {                                   // the single straddling tile
            #pragma unroll
            for (int r = 0; r < 8; ++r) {
                float val = acc[r];
                if (col < QKV_STRIDE) qkvs [(size_t)(rowBase + r) * QKV_STRIDE + col] = val;
                else                  hproj[(size_t)(rowBase + r) * DOUT + (col - QKV_STRIDE)] = val;
            }
        }
    }
}

// ---------------------------------------------------------------------------
// Attention + pooling: one block per 116-node graph (512 blocks, 256 thr).
//   P1: async-stage K rows + raw src ids into LDS.
//   P2: lane=(edge j, head, K-half); 8-wide segment softmax via shfl_xor.
//   P3: aggregate msgs (V via L2-resident scratch) + skip -> h_proj in LDS
//       (reusing K region) + row means e[i].
//   P4: per-graph softmax over e -> alpha_map; scale rows -> h_weighted.
// ---------------------------------------------------------------------------
__global__ void __launch_bounds__(256)
gt_attn_kernel(const float* __restrict__ qkvs, const int* __restrict__ edge_src,
               float* __restrict__ alpha_out, float* __restrict__ hw_out)
{
    extern __shared__ __align__(16) float smem[];
    float* Kl  = smem;                           // NPG * KL_STRIDE (reused as h_proj)
    int*   Sl  = (int*)(Kl + NPG * KL_STRIDE);   // NPG * DEG raw src ids
    float* Al  = (float*)(Sl + NPG * DEG);       // NPG * 16 edge alphas -> node alphas
    float* El  = Al + NPG * 16;                  // NPG row means
    float* Red = El + NPG;                       // {max, sum}

    const int tid   = threadIdx.x;
    const int lane  = tid & 31;
    const int wv    = tid >> 5;
    const int graph = blockIdx.x;
    const int gbase = graph * NPG;

    // ---- Phase 1: async staging ----
    {
        const uint32_t kloff = lds_addr(Kl);
        for (int c = tid; c < NPG * 58; c += 256) {        // 58 x 16B per K row
            int i = c / 58, cc = c - i * 58;
            async_g2l_b128(kloff + i * (KL_STRIDE * 4) + cc * 16,
                           (const char*)(qkvs + (size_t)(gbase + i) * QKV_STRIDE + DOUT) + cc * 16);
        }
        const uint32_t sloff = lds_addr(Sl);
        for (int c = tid; c < (NPG * DEG) / 4; c += 256)   // 232 x 16B of src ids
            async_g2l_b128(sloff + c * 16,
                           (const char*)(edge_src + (size_t)graph * (NPG * DEG)) + c * 16);
        wait_async0();
    }
    __syncthreads();

    // ---- Phase 2: scores + 8-edge softmax per (dst, head) ----
    {
        const int pair = lane & 15;
        const int head = pair & 1;
        const int j    = pair >> 1;
        const int half = lane >> 4;
        for (int i = wv; i < NPG; i += 8) {
            int s = Sl[i * DEG + j] - gbase;
            const float* qrow = qkvs + (size_t)(gbase + i) * QKV_STRIDE + head * DHEAD;
            const float* krow = Kl + s * KL_STRIDE + head * DHEAD;
            float acc = 0.f;
            int d0 = half * 58;
            #pragma unroll 2
            for (int d = d0; d < d0 + 58; ++d) acc = fmaf(qrow[d], krow[d], acc);
            acc += __shfl_xor(acc, 16, 32);
            acc *= 0.09284766908852593f;                  // 1/sqrt(116)
            float mx = acc;
            mx = fmaxf(mx, __shfl_xor(mx, 2, 32));
            mx = fmaxf(mx, __shfl_xor(mx, 4, 32));
            mx = fmaxf(mx, __shfl_xor(mx, 8, 32));
            float ex = expf(acc - mx);
            float sm = ex;
            sm += __shfl_xor(sm, 2, 32);
            sm += __shfl_xor(sm, 4, 32);
            sm += __shfl_xor(sm, 8, 32);
            float alpha = ex / (sm + 1e-16f);
            if (half == 0) Al[i * 16 + pair] = alpha;
        }
    }
    __syncthreads();

    // ---- Phase 3: aggregate + skip; h_proj into LDS (reuse Kl); e[i] ----
    float* HP = Kl;
    for (int i = wv; i < NPG; i += 8) {
        int sj[DEG];
        #pragma unroll
        for (int jj = 0; jj < DEG; ++jj) sj[jj] = Sl[i * DEG + jj] - gbase;
        const float* sproj = hw_out + (size_t)(gbase + i) * DOUT;  // h @ Ws
        float esum = 0.f;
        for (int dd = lane; dd < DOUT; dd += 32) {
            int hd = (dd >= DHEAD) ? 1 : 0;
            float val = sproj[dd];
            #pragma unroll
            for (int jj = 0; jj < DEG; ++jj) {
                float a  = Al[i * 16 + jj * 2 + hd];
                float vv = qkvs[(size_t)(gbase + sj[jj]) * QKV_STRIDE + 2 * DOUT + dd];
                val = fmaf(a, vv, val);
            }
            HP[i * KL_STRIDE + dd] = val;
            esum += val;
        }
        #pragma unroll
        for (int o = 16; o > 0; o >>= 1) esum += __shfl_xor(esum, o, 32);
        if (lane == 0) El[i] = esum * (1.0f / (float)DOUT);
    }
    __syncthreads();

    // ---- Phase 4: per-graph softmax over node means, then scale rows ----
    if (wv == 0) {
        float mx = -3.402823466e38f;
        for (int i = lane; i < NPG; i += 32) mx = fmaxf(mx, El[i]);
        #pragma unroll
        for (int o = 16; o > 0; o >>= 1) mx = fmaxf(mx, __shfl_xor(mx, o, 32));
        float ss = 0.f;
        for (int i = lane; i < NPG; i += 32) ss += expf(El[i] - mx);
        #pragma unroll
        for (int o = 16; o > 0; o >>= 1) ss += __shfl_xor(ss, o, 32);
        if (lane == 0) { Red[0] = mx; Red[1] = ss; }
    }
    __syncthreads();
    {
        float gm = Red[0], gs = Red[1];
        for (int i = tid; i < NPG; i += 256) {
            float a = expf(El[i] - gm) / gs;
            alpha_out[(size_t)graph * NPG + i] = a;
            Al[i] = a;   // reuse: node alphas
        }
    }
    __syncthreads();
    for (int i = wv; i < NPG; i += 8) {
        float a = Al[i];
        size_t ob = (size_t)(gbase + i) * DOUT;
        for (int dd = lane; dd < DOUT; dd += 32)
            hw_out[ob + dd] = a * HP[i * KL_STRIDE + dd];
    }
}

// ---------------------------------------------------------------------------
// Host launcher
// ---------------------------------------------------------------------------
extern "C" void kernel_launch(void* const* d_in, const int* in_sizes, int n_in,
                              void* d_out, int out_size, void* d_ws, size_t ws_size,
                              hipStream_t stream)
{
    (void)in_sizes; (void)n_in; (void)out_size; (void)ws_size;

    const float* h    = (const float*)d_in[0];
    const int*   edge = (const int*)d_in[1];   // [2, E]; row 0 = src
    const float* Wq   = (const float*)d_in[3];
    const float* Wk   = (const float*)d_in[4];
    const float* Wv   = (const float*)d_in[5];
    const float* Ws   = (const float*)d_in[6];

    float* out       = (float*)d_out;
    float* alpha_out = out;                    // [B, 116]
    float* hw_out    = out + (size_t)NTOT;     // [N, 232]

    // Workspace carve (all 16B-aligned): qkvs f32 | h_bf bf16 | Wb bf16
    float*  qkvs = (float*)d_ws;
    __bf16* hb   = (__bf16*)((char*)d_ws + (size_t)NTOT * QKV_STRIDE * sizeof(float));
    __bf16* Wb   = (__bf16*)((char*)hb + (size_t)NTOT * HID * sizeof(__bf16));

    cvt_h_kernel <<<dim3(NTOT / 4), dim3(256), 0, stream>>>(h, hb);
    pack_w_kernel<<<dim3(HID),      dim3(256), 0, stream>>>(Wq, Wk, Wv, Ws, Wb);

    size_t smemA = (size_t)(TILE_M * A_STRIDE + HID * B_STRIDE) * sizeof(__bf16); // ~69 KB
    (void)hipFuncSetAttribute((const void*)gt_gemm_kernel,
                              hipFuncAttributeMaxDynamicSharedMemorySize, (int)smemA);
    dim3 gA((NCOLS + TILE_N - 1) / TILE_N, NTOT / TILE_M);   // 15 x 928
    gt_gemm_kernel<<<gA, dim3(256), smemA, stream>>>(hb, Wb, qkvs, hw_out);

    size_t smemB = (size_t)(NPG * KL_STRIDE + NPG * DEG + NPG * 16 + NPG + 2) * sizeof(float);
    (void)hipFuncSetAttribute((const void*)gt_attn_kernel,
                              hipFuncAttributeMaxDynamicSharedMemorySize, (int)smemB);
    gt_attn_kernel<<<dim3(NGRAPH), dim3(256), smemB, stream>>>(qkvs, edge, alpha_out, hw_out);
}
